// DGCNN_56882546868314
// MI455X (gfx1250) — matrix operand order repaired and checked by reading, not compiled, not run on hardware
//
#include <hip/hip_runtime.h>

// DGCNN forward for MI455X (gfx1250, wave32). All GEMMs via v_wmma_f32_16x16x32_f16.

typedef __attribute__((ext_vector_type(16))) _Float16 v16h;
typedef __attribute__((ext_vector_type(8)))  _Float16 v8h;
typedef __attribute__((ext_vector_type(8)))  float    v8f;

#define DEV __device__ __forceinline__

constexpr int BB = 8, NP = 1024, KNN = 20;
constexpr int NPTS  = BB * NP;        // 8192
constexpr int NROWS = NPTS * KNN;     // 163840
constexpr float EPSB = 1e-5f;

DEV float lrelu(float x) { return x >= 0.f ? x : 0.2f * x; }
DEV float bnv(float v, float mean, float rstd, float g, float b) {
  return (v - mean) * rstd * g + b;
}

// ---------------- weight convert + pad to f16 ----------------
__global__ void k_cvt_w(const float* __restrict__ w, _Float16* __restrict__ o,
                        int O, int C, int Op, int Cp) {
  int i = blockIdx.x * blockDim.x + threadIdx.x;
  if (i >= Op * Cp) return;
  int oo = i / Cp, cc = i % Cp;
  float v = (oo < O && cc < C) ? w[oo * C + cc] : 0.f;
  o[i] = (_Float16)v;
}

// ---------------- (B,3,N) -> (B*N,3) ----------------
__global__ void k_pts_from_x(const float* __restrict__ x, float* __restrict__ pts) {
  int i = blockIdx.x * blockDim.x + threadIdx.x;
  if (i >= NPTS) return;
  int b = i / NP, n = i % NP;
#pragma unroll
  for (int c = 0; c < 3; ++c) pts[i * 3 + c] = x[(b * 3 + c) * NP + n];
}

// ---------------- brute-force kNN (top-20 by -dist), one thread per point ----
__global__ __launch_bounds__(256) void k_knn(const float* __restrict__ pts, int stride,
                                             int C, int* __restrict__ idx) {
  int i = blockIdx.x * blockDim.x + threadIdx.x;
  if (i >= NPTS) return;
  int b = i / NP, n = i % NP;
  const float* base = pts + (size_t)b * NP * stride;
  const float* pn = base + (size_t)n * stride;
  float nn = 0.f;
  for (int c = 0; c < C; ++c) { float v = pn[c]; nn += v * v; }
  float bv[KNN]; int bi[KNN];
#pragma unroll
  for (int k = 0; k < KNN; ++k) { bv[k] = -3.4e38f; bi[k] = 0; }
  for (int m = 0; m < NP; ++m) {
    const float* pm = base + (size_t)m * stride;
    float dot = 0.f, mm = 0.f;
    for (int c = 0; c < C; ++c) { float v = pm[c]; dot += v * pn[c]; mm += v * v; }
    float nd = 2.f * dot - nn - mm;
    if (nd > bv[KNN - 1]) {
      int j = KNN - 1;
      while (j > 0 && bv[j - 1] < nd) { bv[j] = bv[j - 1]; bi[j] = bi[j - 1]; --j; }
      bv[j] = nd; bi[j] = m;
    }
  }
  for (int k = 0; k < KNN; ++k) idx[(size_t)i * KNN + k] = bi[k];
}

// ---------------- edge features [nbr-ctr, ctr] -> f16 A (NROWS x Kp) ----------
__global__ void k_edge_feat(const float* __restrict__ pts, int stride, int C, int Kp,
                            const int* __restrict__ idx, _Float16* __restrict__ A) {
  int i = blockIdx.x * blockDim.x + threadIdx.x;
  if (i >= NROWS * C) return;
  int r = i / C, c = i % C;
  int p = r / KNN;
  int b = p / NP;
  int m = idx[r];
  float ctr = pts[(size_t)p * stride + c];
  float nbr = pts[((size_t)b * NP + m) * stride + c];
  A[(size_t)r * Kp + c]     = (_Float16)(nbr - ctr);
  A[(size_t)r * Kp + C + c] = (_Float16)ctr;
}

// ---------------- WMMA GEMM: Z[M,Npad] = A[M,Kp] * W[Npad,Kp]^T --------------
// wave computes a 16x64 tile; A fragment reused across 4 N-subtiles.
__global__ __launch_bounds__(256) void k_wmma_gemm(const _Float16* __restrict__ A,
                                                   const _Float16* __restrict__ W,
                                                   float* __restrict__ Z,
                                                   int M, int Npad, int Kp) {
  int gw = (int)(((size_t)blockIdx.x * blockDim.x + threadIdx.x) >> 5);
  int lane = threadIdx.x & 31;
  int tilesN = Npad >> 6;
  int tm = gw / tilesN, tn = gw % tilesN;
  if (tm * 16 >= M) return;
  int l15 = lane & 15, hi = lane >> 4;
  v8f acc0 = {}, acc1 = {}, acc2 = {}, acc3 = {};
  const _Float16* arow = A + (size_t)(tm * 16 + l15) * Kp;
  for (int k0 = 0; k0 < Kp; k0 += 32) {
    // A fragment: lanes 0-15 hold K {0..15}; lanes 16-31 hold K {16..31}
    v8h alo = *(const v8h*)(arow + k0 + hi * 16);
    v8h ahi = *(const v8h*)(arow + k0 + hi * 16 + 8);
    v16h a = __builtin_shufflevector(alo, ahi, 0, 1, 2, 3, 4, 5, 6, 7,
                                     8, 9, 10, 11, 12, 13, 14, 15);
    // B fragment: lane holds 16 contiguous K at col = l15, K-half chosen by hi
    const _Float16* wbase = W + (size_t)(tn * 64 + l15) * Kp + k0 + hi * 16;
    v16h b0 = *(const v16h*)(wbase);
    v16h b1 = *(const v16h*)(wbase + (size_t)16 * Kp);
    v16h b2 = *(const v16h*)(wbase + (size_t)32 * Kp);
    v16h b3 = *(const v16h*)(wbase + (size_t)48 * Kp);
    acc0 = __builtin_amdgcn_wmma_f32_16x16x32_f16(false, a, false, b0, (short)0, acc0, false, false);
    acc1 = __builtin_amdgcn_wmma_f32_16x16x32_f16(false, a, false, b1, (short)0, acc1, false, false);
    acc2 = __builtin_amdgcn_wmma_f32_16x16x32_f16(false, a, false, b2, (short)0, acc2, false, false);
    acc3 = __builtin_amdgcn_wmma_f32_16x16x32_f16(false, a, false, b3, (short)0, acc3, false, false);
  }
  // C/D layout: VGPR r holds M = hi*8 + r, N = l15
  float* zrow = Z + (size_t)(tm * 16 + hi * 8) * Npad + (size_t)tn * 64 + l15;
#pragma unroll
  for (int r = 0; r < 8; ++r) {
    zrow[(size_t)r * Npad + 0]  = acc0[r];
    zrow[(size_t)r * Npad + 16] = acc1[r];
    zrow[(size_t)r * Npad + 32] = acc2[r];
    zrow[(size_t)r * Npad + 48] = acc3[r];
  }
}

// ---------------- BN stats: one block per channel -> stats[c]=sum, stats[C+c]=sumsq
__global__ __launch_bounds__(256) void k_bn_stats(const float* __restrict__ Z, int M,
                                                  int Npad, float* __restrict__ stats,
                                                  int C) {
  int c = blockIdx.x;
  float s = 0.f, ss = 0.f;
  for (int r = threadIdx.x; r < M; r += 256) {
    float v = Z[(size_t)r * Npad + c];
    s += v; ss += v * v;
  }
  __shared__ float sh0[256], sh1[256];
  sh0[threadIdx.x] = s; sh1[threadIdx.x] = ss; __syncthreads();
  for (int o = 128; o > 0; o >>= 1) {
    if (threadIdx.x < o) { sh0[threadIdx.x] += sh0[threadIdx.x + o];
                           sh1[threadIdx.x] += sh1[threadIdx.x + o]; }
    __syncthreads();
  }
  if (threadIdx.x == 0) { stats[c] = sh0[0]; stats[C + c] = sh1[0]; }
}

// ---------------- BN + lrelu elementwise -> f16 A (stride C) -----------------
__global__ void k_bn_lrelu_toA(const float* __restrict__ Z, int M, int Npad, int C,
                               const float* __restrict__ stats, const float* __restrict__ g,
                               const float* __restrict__ bb, _Float16* __restrict__ A) {
  int i = blockIdx.x * blockDim.x + threadIdx.x;
  if (i >= M * C) return;
  int r = i / C, c = i % C;
  float mean = stats[c] / M;
  float rstd = rsqrtf(stats[C + c] / M - mean * mean + EPSB);
  A[i] = (_Float16)lrelu(bnv(Z[(size_t)r * Npad + c], mean, rstd, g[c], bb[c]));
}

// ---------------- BN + lrelu + max over K -> f16 A (NPTS x C) ----------------
__global__ void k_bn_lrelu_maxk_toA(const float* __restrict__ Z, int Npad, int C,
                                    const float* __restrict__ stats, const float* __restrict__ g,
                                    const float* __restrict__ bb, _Float16* __restrict__ A) {
  int i = blockIdx.x * blockDim.x + threadIdx.x;
  if (i >= NPTS * C) return;
  int p = i / C, c = i % C;
  float mean = stats[c] / NROWS;
  float rstd = rsqrtf(stats[C + c] / NROWS - mean * mean + EPSB);
  float gm = g[c], bm = bb[c], mx = -3.4e38f;
  for (int k = 0; k < KNN; ++k)
    mx = fmaxf(mx, lrelu(bnv(Z[((size_t)p * KNN + k) * Npad + c], mean, rstd, gm, bm)));
  A[(size_t)p * C + c] = (_Float16)mx;
}

// ---------------- BN + lrelu + max over K -> f32 concat slice ----------------
__global__ void k_bn_lrelu_maxk_toX(const float* __restrict__ Z, int Npad, int C,
                                    const float* __restrict__ stats, const float* __restrict__ g,
                                    const float* __restrict__ bb, float* __restrict__ xcat,
                                    int colOff) {
  int i = blockIdx.x * blockDim.x + threadIdx.x;
  if (i >= NPTS * C) return;
  int p = i / C, c = i % C;
  float mean = stats[c] / NROWS;
  float rstd = rsqrtf(stats[C + c] / NROWS - mean * mean + EPSB);
  float gm = g[c], bm = bb[c], mx = -3.4e38f;
  for (int k = 0; k < KNN; ++k)
    mx = fmaxf(mx, lrelu(bnv(Z[((size_t)p * KNN + k) * Npad + c], mean, rstd, gm, bm)));
  xcat[(size_t)p * 512 + colOff + c] = mx;
}

// ---------------- BN + lrelu + max over N -> f16 A (16 x C, rows 8..15 pre-zeroed)
__global__ void k_bn_lrelu_maxn_toA(const float* __restrict__ Z, int Npad, int C,
                                    const float* __restrict__ stats, const float* __restrict__ g,
                                    const float* __restrict__ bb, _Float16* __restrict__ A) {
  int i = blockIdx.x * blockDim.x + threadIdx.x;
  if (i >= BB * C) return;
  int b = i / C, c = i % C;
  float mean = stats[c] / NPTS;
  float rstd = rsqrtf(stats[C + c] / NPTS - mean * mean + EPSB);
  float gm = g[c], bm = bb[c], mx = -3.4e38f;
  for (int n = 0; n < NP; ++n)
    mx = fmaxf(mx, lrelu(bnv(Z[((size_t)b * NP + n) * Npad + c], mean, rstd, gm, bm)));
  A[(size_t)b * C + c] = (_Float16)mx;
}

// ---------------- FC epilogue: +bias, BN over batch (8 rows), lrelu -> f16 A --
__global__ void k_fc_head(const float* __restrict__ Z, int Npad, const float* __restrict__ bias,
                          const float* __restrict__ g, const float* __restrict__ bb, int C,
                          _Float16* __restrict__ A) {
  int c = blockIdx.x * blockDim.x + threadIdx.x;
  if (c >= C) return;
  float v[BB]; float s = 0.f;
#pragma unroll
  for (int r = 0; r < BB; ++r) { v[r] = Z[(size_t)r * Npad + c] + bias[c]; s += v[r]; }
  float mean = s / BB, ss = 0.f;
#pragma unroll
  for (int r = 0; r < BB; ++r) { float d = v[r] - mean; ss += d * d; }
  float rstd = rsqrtf(ss / BB + EPSB);
#pragma unroll
  for (int r = 0; r < BB; ++r)
    A[(size_t)r * C + c] = (_Float16)lrelu((v[r] - mean) * rstd * g[c] + bb[c]);
}

// ---------------- t matrix: Z(16x64, first 9 cols) + bias -> t (B,9) ---------
__global__ void k_tmat(const float* __restrict__ Z, int Npad, const float* __restrict__ bias,
                       float* __restrict__ t) {
  int i = threadIdx.x;
  if (i >= BB * 9) return;
  int b = i / 9, j = i % 9;
  t[b * 9 + j] = Z[(size_t)b * Npad + j] + bias[j];
}

// ---------------- apply 3x3 transform: pts_t = t @ x -------------------------
__global__ void k_transform(const float* __restrict__ x, const float* __restrict__ t,
                            float* __restrict__ pts) {
  int i = blockIdx.x * blockDim.x + threadIdx.x;
  if (i >= NPTS) return;
  int b = i / NP, n = i % NP;
  float p0 = x[(b * 3 + 0) * NP + n];
  float p1 = x[(b * 3 + 1) * NP + n];
  float p2 = x[(b * 3 + 2) * NP + n];
  const float* tb = t + b * 9;
#pragma unroll
  for (int c = 0; c < 3; ++c)
    pts[(size_t)i * 3 + c] = tb[c * 3 + 0] * p0 + tb[c * 3 + 1] * p1 + tb[c * 3 + 2] * p2;
}

// ---------------- f32 -> f16 copy --------------------------------------------
__global__ void k_f32_to_f16(const float* __restrict__ src, _Float16* __restrict__ dst, int n) {
  int i = blockIdx.x * blockDim.x + threadIdx.x;
  if (i < n) dst[i] = (_Float16)src[i];
}

// ---------------- final logits -----------------------------------------------
__global__ void k_final(const float* __restrict__ Z, int Npad, const float* __restrict__ bias,
                        float* __restrict__ out) {
  int i = blockIdx.x * blockDim.x + threadIdx.x;
  if (i >= BB * 40) return;
  int b = i / 40, c = i % 40;
  out[i] = Z[(size_t)b * Npad + c] + bias[c];
}

// =============================================================================
extern "C" void kernel_launch(void* const* d_in, const int* in_sizes, int n_in,
                              void* d_out, int out_size, void* d_ws, size_t ws_size,
                              hipStream_t stream) {
  (void)in_sizes; (void)n_in; (void)out_size; (void)ws_size;
  const float* x = (const float*)d_in[0];
  auto P = [&](int i) { return (const float*)d_in[i]; };

  // ---- workspace carve-out (all 256B aligned) ----
  char* ws = (char*)d_ws;
  size_t off = 0;
  auto alloc = [&](size_t bytes) -> char* {
    char* p = ws + off;
    off = (off + bytes + 255) & ~(size_t)255;
    return p;
  };
  _Float16* A    = (_Float16*)alloc((size_t)NROWS * 256 * sizeof(_Float16)); //  80 MB
  float*    Z    = (float*)   alloc((size_t)NROWS * 256 * sizeof(float));    // 160 MB
  _Float16* WH   = (_Float16*)alloc((size_t)2101248 * sizeof(_Float16));     //   4 MB
  int*      idx  = (int*)     alloc((size_t)NROWS * sizeof(int));
  float*    pts0 = (float*)   alloc((size_t)NPTS * 3 * sizeof(float));
  float*    ptst = (float*)   alloc((size_t)NPTS * 3 * sizeof(float));
  float*    xcat = (float*)   alloc((size_t)NPTS * 512 * sizeof(float));     //  16 MB
  float*    stats= (float*)   alloc(4096 * sizeof(float));
  float*    tmat = (float*)   alloc(256 * sizeof(float));

  auto g1d = [](int n) { return dim3((n + 255) / 256); };

  // ---- convert + pad all weights to f16 ----
  size_t woff = 0;
  auto cvt = [&](int argi, int O, int C, int Op, int Cp) -> const _Float16* {
    _Float16* dst = WH + woff;
    int n = Op * Cp;
    k_cvt_w<<<g1d(n), 256, 0, stream>>>(P(argi), dst, O, C, Op, Cp);
    woff += (size_t)n;
    return dst;
  };
  const _Float16* w_tc1 = cvt(1,    64,    6,   64,   32);
  const _Float16* w_tc2 = cvt(4,   128,   64,  128,   64);
  const _Float16* w_tc3 = cvt(7,  1024,  128, 1024,  128);
  const _Float16* w_tf1 = cvt(10,  512, 1024,  512, 1024);
  const _Float16* w_tf2 = cvt(14,  256,  512,  256,  512);
  const _Float16* w_to  = cvt(18,    9,  256,   64,  256);
  const _Float16* w_c1  = cvt(20,   64,    6,   64,   32);
  const _Float16* w_c2  = cvt(23,   64,  128,   64,  128);
  const _Float16* w_c3  = cvt(26,  128,  128,  128,  128);
  const _Float16* w_c4  = cvt(29,  256,  256,  256,  256);
  const _Float16* w_c5  = cvt(32, 1024,  512, 1024,  512);
  const _Float16* w_f1  = cvt(35,  512, 1024,  512, 1024);
  const _Float16* w_f2  = cvt(39,  256,  512,  256,  512);
  const _Float16* w_f3  = cvt(43,   40,  256,   64,  256);

  auto gemm = [&](const _Float16* Am, const _Float16* Wm, float* Zm, int M, int Npad, int Kp) {
    int waves = (M / 16) * (Npad / 64);
    k_wmma_gemm<<<dim3((waves + 7) / 8), 256, 0, stream>>>(Am, Wm, Zm, M, Npad, Kp);
  };
  auto bstats = [&](const float* Zm, int M, int Npad, int C) {
    k_bn_stats<<<dim3(C), 256, 0, stream>>>(Zm, M, Npad, stats, C);
  };

  // ---- (B,3,N) -> (B*N,3) ----
  k_pts_from_x<<<g1d(NPTS), 256, 0, stream>>>(x, pts0);

  // ================= transform_net =================
  k_knn<<<g1d(NPTS), 256, 0, stream>>>(pts0, 3, 3, idx);
  hipMemsetAsync(A, 0, (size_t)NROWS * 32 * sizeof(_Float16), stream);
  k_edge_feat<<<g1d(NROWS * 3), 256, 0, stream>>>(pts0, 3, 3, 32, idx, A);
  gemm(A, w_tc1, Z, NROWS, 64, 32);
  bstats(Z, NROWS, 64, 64);
  k_bn_lrelu_toA<<<g1d(NROWS * 64), 256, 0, stream>>>(Z, NROWS, 64, 64, stats, P(2), P(3), A);
  gemm(A, w_tc2, Z, NROWS, 128, 64);
  bstats(Z, NROWS, 128, 128);
  k_bn_lrelu_maxk_toA<<<g1d(NPTS * 128), 256, 0, stream>>>(Z, 128, 128, stats, P(5), P(6), A);
  gemm(A, w_tc3, Z, NPTS, 1024, 128);
  bstats(Z, NPTS, 1024, 1024);
  hipMemsetAsync(A, 0, (size_t)16 * 1024 * sizeof(_Float16), stream);
  k_bn_lrelu_maxn_toA<<<g1d(BB * 1024), 256, 0, stream>>>(Z, 1024, 1024, stats, P(8), P(9), A);
  gemm(A, w_tf1, Z, 16, 512, 1024);
  hipMemsetAsync(A, 0, (size_t)16 * 512 * sizeof(_Float16), stream);
  k_fc_head<<<g1d(512), 256, 0, stream>>>(Z, 512, P(11), P(12), P(13), 512, A);
  gemm(A, w_tf2, Z, 16, 256, 512);
  hipMemsetAsync(A, 0, (size_t)16 * 256 * sizeof(_Float16), stream);
  k_fc_head<<<g1d(256), 256, 0, stream>>>(Z, 256, P(15), P(16), P(17), 256, A);
  gemm(A, w_to, Z, 16, 64, 256);
  k_tmat<<<dim3(1), 128, 0, stream>>>(Z, 64, P(19), tmat);
  k_transform<<<g1d(NPTS), 256, 0, stream>>>(x, tmat, ptst);

  // ================= edge-conv chain =================
  auto edge_conv = [&](const float* in, int stride, int C, const _Float16* Wm,
                       int Npad, int Cout, const float* g, const float* bb, int colOff) {
    k_knn<<<g1d(NPTS), 256, 0, stream>>>(in, stride, C, idx);
    int Kp = (2 * C < 32) ? 32 : 2 * C;
    if (Kp != 2 * C)
      hipMemsetAsync(A, 0, (size_t)NROWS * Kp * sizeof(_Float16), stream);
    k_edge_feat<<<g1d(NROWS * C), 256, 0, stream>>>(in, stride, C, Kp, idx, A);
    gemm(A, Wm, Z, NROWS, Npad, Kp);
    bstats(Z, NROWS, Npad, Cout);
    k_bn_lrelu_maxk_toX<<<g1d(NPTS * Cout), 256, 0, stream>>>(Z, Npad, Cout, stats, g, bb,
                                                              xcat, colOff);
  };
  edge_conv(ptst,       3,   3, w_c1,  64,  64, P(21), P(22),   0);  // x1 -> cols 0..63
  edge_conv(xcat + 0,  512,  64, w_c2,  64,  64, P(24), P(25),  64); // x2 -> cols 64..127
  edge_conv(xcat + 64, 512,  64, w_c3, 128, 128, P(27), P(28), 128); // x3 -> cols 128..255
  edge_conv(xcat + 128,512, 128, w_c4, 256, 256, P(30), P(31), 256); // x4 -> cols 256..511

  // ================= c5 + global max + FC head =================
  k_f32_to_f16<<<g1d(NPTS * 512), 256, 0, stream>>>(xcat, A, NPTS * 512);
  gemm(A, w_c5, Z, NPTS, 1024, 512);
  bstats(Z, NPTS, 1024, 1024);
  hipMemsetAsync(A, 0, (size_t)16 * 1024 * sizeof(_Float16), stream);
  k_bn_lrelu_maxn_toA<<<g1d(BB * 1024), 256, 0, stream>>>(Z, 1024, 1024, stats, P(33), P(34), A);
  gemm(A, w_f1, Z, 16, 512, 1024);
  hipMemsetAsync(A, 0, (size_t)16 * 512 * sizeof(_Float16), stream);
  k_fc_head<<<g1d(512), 256, 0, stream>>>(Z, 512, P(36), P(37), P(38), 512, A);
  gemm(A, w_f2, Z, 16, 256, 512);
  hipMemsetAsync(A, 0, (size_t)16 * 256 * sizeof(_Float16), stream);
  k_fc_head<<<g1d(256), 256, 0, stream>>>(Z, 256, P(40), P(41), P(42), 256, A);
  gemm(A, w_f3, Z, 16, 64, 256);
  k_final<<<g1d(BB * 40), 256, 0, stream>>>(Z, 64, P(44), (float*)d_out);
}
